// PoGNet_77919296684491
// MI455X (gfx1250) — matrix operand order repaired and checked by reading, not compiled, not run on hardware
//
#include <hip/hip_runtime.h>
#include <hip/hip_bf16.h>

// ---- problem constants (from reference) ----
#define NB    64      // batch
#define NPp   32      // pieces
#define NSn   72      // nodes
#define NH    4       // heads
#define NF    64      // feat/head
#define HD    256     // hidden = NH*NF
#define NL    4       // layers
#define NCARD 110
#define KCTX  112     // CS*CE = 7*16
#define KCTXP 128     // padded K for ctx GEMM
#define ROWS  (NB*NSn)   // 4608 flattened node rows (divisible by 32)

typedef __attribute__((ext_vector_type(16))) _Float16 v16h;
typedef __attribute__((ext_vector_type(8)))  float    v8f;

// ---------------------------------------------------------------------------
// f16 WMMA GEMM: C(MxN) = A(MxK,f16,row-major) * Bt(NxK,f16 = B transposed)
//   + optional bias (len N) + optional residual (MxN f32).
// Assumes: M % 32 == 0, N % 128 == 0 (always 256 here), K % 32 == 0.
// 4 waves / block; each wave owns a 32x32 C tile: 2 A frags x 2 B frags ->
// 4 accumulators, so each k-step is 8 contiguous b128 loads + 4 WMMAs.
// ---------------------------------------------------------------------------
__global__ __launch_bounds__(128) void wmma_gemm_kernel(
    const _Float16* __restrict__ A, const _Float16* __restrict__ Bt,
    const float* __restrict__ bias, const float* __restrict__ residual,
    float* __restrict__ C, int M, int N, int K)
{
    const int lane = threadIdx.x & 31;
    const int wave = threadIdx.x >> 5;
    const int tM   = blockIdx.y;                    // 32-row band
    const int tN0  = (blockIdx.x * 4 + wave) * 2;   // first of two 16-col tiles

    // A fragment (16-bit A 16x32), same k order as B:
    //  lanes 0-15: row = lane&15, halves 0..15 -> K 0..15
    //  lanes 16-31: same rows, K 16..31
    const int rowA0 = tM * 32 + (lane & 15);
    const int rowA1 = rowA0 + 16;
    const int kA    = (lane < 16) ? 0 : 16;
    // B fragment: col = lane&15; lanes 0-15 hold K 0..15, lanes 16-31 K 16..31.
    // With Bt (col-major B), each fragment is 16 consecutive halves.
    const int col0  = tN0 * 16 + (lane & 15);
    const int col1  = col0 + 16;
    const int kB    = (lane < 16) ? 0 : 16;

    const _Float16* __restrict__ ap0 = A  + (size_t)rowA0 * K + kA;
    const _Float16* __restrict__ ap1 = A  + (size_t)rowA1 * K + kA;
    const _Float16* __restrict__ bp0 = Bt + (size_t)col0  * K + kB;
    const _Float16* __restrict__ bp1 = Bt + (size_t)col1  * K + kB;

    v8f acc00 = {}, acc01 = {}, acc10 = {}, acc11 = {};
    for (int k0 = 0; k0 < K; k0 += 32) {
        v16h af0, af1, bf0, bf1;
        #pragma unroll
        for (int i = 0; i < 8; ++i) {
            af0[i] = ap0[k0 + i]; af0[8 + i] = ap0[k0 + 8 + i];
            af1[i] = ap1[k0 + i]; af1[8 + i] = ap1[k0 + 8 + i];
        }
        #pragma unroll
        for (int i = 0; i < 16; ++i) { bf0[i] = bp0[k0 + i]; bf1[i] = bp1[k0 + i]; }
        acc00 = __builtin_amdgcn_wmma_f32_16x16x32_f16(false, af0, false, bf0,
                                                       (short)0, acc00, false, false);
        acc01 = __builtin_amdgcn_wmma_f32_16x16x32_f16(false, af0, false, bf1,
                                                       (short)0, acc01, false, false);
        acc10 = __builtin_amdgcn_wmma_f32_16x16x32_f16(false, af1, false, bf0,
                                                       (short)0, acc10, false, false);
        acc11 = __builtin_amdgcn_wmma_f32_16x16x32_f16(false, af1, false, bf1,
                                                       (short)0, acc11, false, false);
    }

    // D layout: reg r -> row r (lanes 0-15) / row r+8 (lanes 16-31), col = lane&15
    const int rb0 = tM * 32 + ((lane < 16) ? 0 : 8);
    const int rb1 = rb0 + 16;
    const float b0 = bias ? bias[col0] : 0.0f;
    const float b1 = bias ? bias[col1] : 0.0f;
    #pragma unroll
    for (int r = 0; r < 8; ++r) {
        const size_t row0 = (size_t)(rb0 + r);
        const size_t row1 = (size_t)(rb1 + r);
        float v00 = acc00[r] + b0, v01 = acc01[r] + b1;
        float v10 = acc10[r] + b0, v11 = acc11[r] + b1;
        if (residual) {
            v00 += residual[row0 * N + col0];
            v01 += residual[row0 * N + col1];
            v10 += residual[row1 * N + col0];
            v11 += residual[row1 * N + col1];
        }
        C[row0 * N + col0] = v00;
        C[row0 * N + col1] = v01;
        C[row1 * N + col0] = v10;
        C[row1 * N + col1] = v11;
    }
}

// ---------------------------------------------------------------------------
// Weight repack: batch of `count` KxN f32 matrices -> NxK f16 (transposed)
// ---------------------------------------------------------------------------
__global__ void cvt_f16_T_kernel(const float* __restrict__ src,
                                 _Float16* __restrict__ dst,
                                 int K, int N, int total)
{
    int i = blockIdx.x * blockDim.x + threadIdx.x;   // over count*N*K outputs
    if (i >= total) return;
    int m = i / (N * K);
    int r = i - m * (N * K);
    int n = r / K;
    int k = r - n * K;
    dst[i] = (_Float16)src[(size_t)m * K * N + (size_t)k * N + n];
}

// ctx_W (112x256) -> transposed padded f16 (256x128), zeros for k >= 112
__global__ void ctxw_padT_kernel(const float* __restrict__ src,
                                 _Float16* __restrict__ dst)
{
    int i = blockIdx.x * blockDim.x + threadIdx.x;   // 256*128
    if (i >= HD * KCTXP) return;
    int n = i / KCTXP, k = i % KCTXP;
    dst[i] = (k < KCTX) ? (_Float16)src[k * HD + n] : (_Float16)0.0f;
}

// card_context (64x112) -> padded f16 A (64x128), row-major
__global__ void ctxa_pad_kernel(const float* __restrict__ src,
                                _Float16* __restrict__ dst)
{
    int i = blockIdx.x * blockDim.x + threadIdx.x;   // 64*128
    if (i >= NB * KCTXP) return;
    int b = i / KCTXP, k = i % KCTXP;
    dst[i] = (k < KCTX) ? (_Float16)src[b * KCTX + k] : (_Float16)0.0f;
}

// spatial_obs (B,NP,NS) -> embedding A matrix (B*NS, NP) f16 (transpose p<->s)
__global__ void embedA_kernel(const float* __restrict__ so,
                              _Float16* __restrict__ A)
{
    int i = blockIdx.x * blockDim.x + threadIdx.x;   // ROWS*NPp
    if (i >= ROWS * NPp) return;
    int p = i & (NPp - 1);
    int row = i >> 5;
    int b = row / NSn, s = row % NSn;
    A[i] = (_Float16)so[((size_t)b * NPp + p) * NSn + s];
}

// ---------------------------------------------------------------------------
// LayerNorm of ctx projection + ReLU -> g (f16); one block per batch row
// ---------------------------------------------------------------------------
__global__ __launch_bounds__(256) void ln_g_kernel(
    const float* __restrict__ gl, const float* __restrict__ s,
    const float* __restrict__ bb, _Float16* __restrict__ g)
{
    int b = blockIdx.x, t = threadIdx.x;
    __shared__ float red[256];
    float v = gl[(size_t)b * HD + t];
    red[t] = v; __syncthreads();
    for (int o = 128; o > 0; o >>= 1) { if (t < o) red[t] += red[t + o]; __syncthreads(); }
    float mean = red[0] * (1.0f / HD);
    __syncthreads();
    float d = v - mean;
    red[t] = d * d; __syncthreads();
    for (int o = 128; o > 0; o >>= 1) { if (t < o) red[t] += red[t + o]; __syncthreads(); }
    float var = red[0] * (1.0f / HD);
    float r = rsqrtf(var + 1e-6f);
    float hv = d * r * s[t] + bb[t];
    g[(size_t)b * HD + t] = (_Float16)(hv > 0.0f ? hv : 0.0f);
}

// LayerNorm of x rows (+ broadcast ctx projection) -> h (f16); block per node row
__global__ __launch_bounds__(256) void ln_h_kernel(
    const float* __restrict__ x, const float* __restrict__ s,
    const float* __restrict__ bb, const float* __restrict__ gproj,
    _Float16* __restrict__ h)
{
    int row = blockIdx.x, t = threadIdx.x;
    int b = row / NSn;
    __shared__ float red[256];
    float v = x[(size_t)row * HD + t];
    red[t] = v; __syncthreads();
    for (int o = 128; o > 0; o >>= 1) { if (t < o) red[t] += red[t + o]; __syncthreads(); }
    float mean = red[0] * (1.0f / HD);
    __syncthreads();
    float d = v - mean;
    red[t] = d * d; __syncthreads();
    for (int o = 128; o > 0; o >>= 1) { if (t < o) red[t] += red[t + o]; __syncthreads(); }
    float var = red[0] * (1.0f / HD);
    float r = rsqrtf(var + 1e-6f);
    float hv = d * r * s[t] + bb[t] + gproj[(size_t)b * HD + t];
    h[(size_t)row * HD + t] = (_Float16)hv;
}

// ---------------------------------------------------------------------------
// GATv2 attention: one block per (b, i) destination node.
//   scores[j,h] = Wa[h]. leakyrelu(hl[i,h,:] + hr[j,h,:]); masked softmax over j;
//   hg[i,h,f]  = sum_j attn[j,h] * hr[j,h,f]   (stored f16 for op GEMM)
// ---------------------------------------------------------------------------
__global__ __launch_bounds__(256) void attn_kernel(
    const float* __restrict__ hl, const float* __restrict__ hr,
    const float* __restrict__ adj, const float* __restrict__ Wa,
    _Float16* __restrict__ hg)
{
    int bi = blockIdx.x;                 // b*NSn + i
    int b = bi / NSn, i = bi % NSn;
    int t = threadIdx.x;
    __shared__ float hl_s[HD], wa_s[HD], sc_s[NSn * NH];

    hl_s[t] = hl[(size_t)bi * HD + t];
    wa_s[t] = Wa[t];
    __syncthreads();

    for (int p = t; p < NSn * NH; p += 256) {
        int j = p >> 2, hh = p & 3;
        const float* hrj = hr + ((size_t)(b * NSn + j) * HD + hh * NF);
        const float* hli = hl_s + hh * NF;
        const float* wah = wa_s + hh * NF;
        float sc = 0.0f;
        #pragma unroll 8
        for (int f = 0; f < NF; ++f) {
            float v = hli[f] + hrj[f];
            v = (v > 0.0f) ? v : 0.2f * v;       // leaky_relu(.,0.2)
            sc += wah[f] * v;
        }
        sc_s[p] = (adj[i * NSn + j] > 0.0f) ? sc : -1e9f;
    }
    __syncthreads();

    if (t < NH) {  // per-head masked softmax over 72 neighbors
        float mx = -3.4e38f;
        for (int j = 0; j < NSn; ++j) mx = fmaxf(mx, sc_s[j * NH + t]);
        float sum = 0.0f;
        for (int j = 0; j < NSn; ++j) {
            float e = __expf(sc_s[j * NH + t] - mx);
            sc_s[j * NH + t] = e; sum += e;
        }
        float inv = 1.0f / sum;
        for (int j = 0; j < NSn; ++j) sc_s[j * NH + t] *= inv;
    }
    __syncthreads();

    {   // aggregate: t = hh*64 + f
        int hh = t >> 6;
        float acc = 0.0f;
        for (int j = 0; j < NSn; ++j)
            acc += sc_s[j * NH + hh] * hr[(size_t)(b * NSn + j) * HD + t];
        hg[(size_t)bi * HD + t] = (_Float16)acc;
    }
}

// mean over nodes: ge[b,d] = mean_s x[b,s,d]
__global__ __launch_bounds__(256) void mean_kernel(const float* __restrict__ x,
                                                   float* __restrict__ ge)
{
    int b = blockIdx.x, t = threadIdx.x;
    float s = 0.0f;
    for (int n = 0; n < NSn; ++n) s += x[((size_t)b * NSn + n) * HD + t];
    ge[(size_t)b * HD + t] = s * (1.0f / NSn);
}

// output heads; one block per batch row; writes d_out in tuple-concat order
__global__ __launch_bounds__(256) void heads_kernel(
    const float* __restrict__ ge,
    const float* __restrict__ card_W, const float* __restrict__ card_b,
    const float* __restrict__ at_W,   const float* __restrict__ at_b,
    const float* __restrict__ src_W,  const float* __restrict__ src_b,
    const float* __restrict__ tgt_W,  const float* __restrict__ tgt_b,
    const float* __restrict__ v1_W,   const float* __restrict__ v1_b,
    const float* __restrict__ v2_W,   const float* __restrict__ v2_b,
    float* __restrict__ out)
{
    const int b = blockIdx.x, t = threadIdx.x;
    __shared__ float ge_s[HD];
    __shared__ float v1_s[128];
    ge_s[t] = ge[(size_t)b * HD + t];
    __syncthreads();

    const int OFF_AT  = NB * NCARD;              // 7040
    const int OFF_SRC = OFF_AT + NB * 3;         // 7232
    const int OFF_TGT = OFF_SRC + NB * NSn;      // 11840
    const int OFF_VAL = OFF_TGT + NB * NSn;      // 16448

    if (t < NCARD) {
        float a = card_b[t];
        for (int d = 0; d < HD; ++d) a += ge_s[d] * card_W[d * NCARD + t];
        out[b * NCARD + t] = a;
    }
    if (t < 3) {
        float a = at_b[t];
        for (int d = 0; d < HD; ++d) a += ge_s[d] * at_W[d * 3 + t];
        out[OFF_AT + b * 3 + t] = a;
    }
    if (t >= 128 && t < 128 + NSn) {
        int u = t - 128;
        float a = src_b[u];
        for (int d = 0; d < HD; ++d) a += ge_s[d] * src_W[d * NSn + u];
        out[OFF_SRC + b * NSn + u] = a;
    }
    if (t >= 184 && t < 184 + NSn) {
        int u = t - 184;
        float a = tgt_b[u];
        for (int d = 0; d < HD; ++d) a += ge_s[d] * tgt_W[d * NSn + u];
        out[OFF_TGT + b * NSn + u] = a;
    }
    if (t < 128) {
        float a = v1_b[t];
        for (int d = 0; d < HD; ++d) a += ge_s[d] * v1_W[d * 128 + t];
        v1_s[t] = (a > 0.0f) ? a : 0.0f;
    }
    __syncthreads();
    if (t == 0) {
        float a = v2_b[0];
        for (int d = 0; d < 128; ++d) a += v1_s[d] * v2_W[d];
        out[OFF_VAL + b] = tanhf(a);
    }
}

// ---------------------------------------------------------------------------
extern "C" void kernel_launch(void* const* d_in, const int* in_sizes, int n_in,
                              void* d_out, int out_size, void* d_ws, size_t ws_size,
                              hipStream_t stream)
{
    (void)in_sizes; (void)n_in; (void)out_size; (void)ws_size;
    const float* spatial = (const float*)d_in[0];
    const float* cardctx = (const float*)d_in[1];
    const float* adj     = (const float*)d_in[2];
    const float* ctx_W   = (const float*)d_in[3];
    const float* ctx_b   = (const float*)d_in[4];
    const float* lnc_s   = (const float*)d_in[5];
    const float* lnc_b   = (const float*)d_in[6];
    const float* in_W    = (const float*)d_in[7];
    const float* in_b    = (const float*)d_in[8];
    const float* lnp_s   = (const float*)d_in[9];
    const float* lnp_b   = (const float*)d_in[10];
    const float* cp_W    = (const float*)d_in[11];
    const float* cp_b    = (const float*)d_in[12];
    const float* Wl      = (const float*)d_in[13];
    const float* Wr      = (const float*)d_in[14];
    const float* Wa      = (const float*)d_in[15];
    const float* op_W    = (const float*)d_in[16];
    const float* op_b    = (const float*)d_in[17];
    const float* card_W  = (const float*)d_in[18];
    const float* card_b  = (const float*)d_in[19];
    const float* at_W    = (const float*)d_in[20];
    const float* at_b    = (const float*)d_in[21];
    const float* src_W   = (const float*)d_in[22];
    const float* src_b   = (const float*)d_in[23];
    const float* tgt_W   = (const float*)d_in[24];
    const float* tgt_b   = (const float*)d_in[25];
    const float* v1_W    = (const float*)d_in[26];
    const float* v1_b    = (const float*)d_in[27];
    const float* v2_W    = (const float*)d_in[28];
    const float* v2_b    = (const float*)d_in[29];
    float* out = (float*)d_out;

    // ---- workspace carve-up ----
    char* ws = (char*)d_ws;
    size_t off = 0;
    auto take = [&](size_t bytes) { size_t o = off; off += (bytes + 255) & ~(size_t)255; return (void*)(ws + o); };

    _Float16* inWt_h  = (_Float16*)take((size_t)HD * NPp * 2);        // 256x32
    _Float16* cpWt_h  = (_Float16*)take((size_t)NL * HD * HD * 2);    // 4x 256x256 (T)
    _Float16* WlWt_h  = (_Float16*)take((size_t)NL * HD * HD * 2);
    _Float16* WrWt_h  = (_Float16*)take((size_t)NL * HD * HD * 2);
    _Float16* opWt_h  = (_Float16*)take((size_t)NL * HD * HD * 2);
    _Float16* ctxWt_h = (_Float16*)take((size_t)HD * KCTXP * 2);      // 256x128 (T,pad)
    _Float16* Actx_h  = (_Float16*)take((size_t)NB * KCTXP * 2);
    _Float16* Aemb_h  = (_Float16*)take((size_t)ROWS * NPp * 2);
    _Float16* g_h     = (_Float16*)take((size_t)NB * HD * 2);
    _Float16* h_h     = (_Float16*)take((size_t)ROWS * HD * 2);
    _Float16* hg_h    = (_Float16*)take((size_t)ROWS * HD * 2);
    float*    x_f     = (float*)take((size_t)ROWS * HD * 4);
    float*    hl_f    = (float*)take((size_t)ROWS * HD * 4);
    float*    hr_f    = (float*)take((size_t)ROWS * HD * 4);
    float*    gl_f    = (float*)take((size_t)NB * HD * 4);
    float*    gproj   = (float*)take((size_t)NB * HD * 4);
    float*    ge_f    = (float*)take((size_t)NB * HD * 4);

    auto cvtT = [&](const float* s, _Float16* d, int K, int N, int count) {
        int total = count * K * N;
        cvt_f16_T_kernel<<<(total + 255) / 256, 256, 0, stream>>>(s, d, K, N, total);
    };

    // ---- weight repack (transpose + f16) ----
    cvtT(in_W, inWt_h, NPp, HD, 1);
    cvtT(cp_W, cpWt_h, HD, HD, NL);
    cvtT(Wl,   WlWt_h, HD, HD, NL);
    cvtT(Wr,   WrWt_h, HD, HD, NL);
    cvtT(op_W, opWt_h, HD, HD, NL);
    ctxw_padT_kernel<<<(HD * KCTXP + 255) / 256, 256, 0, stream>>>(ctx_W, ctxWt_h);
    ctxa_pad_kernel<<<(NB * KCTXP + 255) / 256, 256, 0, stream>>>(cardctx, Actx_h);
    embedA_kernel<<<(ROWS * NPp + 255) / 256, 256, 0, stream>>>(spatial, Aemb_h);

    auto gemm = [&](const _Float16* A, const _Float16* Bt, const float* bias,
                    const float* res, float* C, int M, int N, int K) {
        dim3 grid(N / 128, M / 32);   // wave = 32x32 tile, block = 32x128
        wmma_gemm_kernel<<<grid, 128, 0, stream>>>(A, Bt, bias, res, C, M, N, K);
    };

    // ---- context branch: g = relu(LN(ctx @ ctx_W + b)) ----
    gemm(Actx_h, ctxWt_h, ctx_b, nullptr, gl_f, NB, HD, KCTXP);
    ln_g_kernel<<<NB, 256, 0, stream>>>(gl_f, lnc_s, lnc_b, g_h);

    // ---- embedding: x[b,s,:] = sum_p so[b,p,s] * in_W[p,:] + in_b ----
    gemm(Aemb_h, inWt_h, in_b, nullptr, x_f, ROWS, HD, NPp);

    // ---- GNN layers ----
    for (int l = 0; l < NL; ++l) {
        const size_t wo = (size_t)l * HD * HD;
        gemm(g_h, cpWt_h + wo, cp_b + l * HD, nullptr, gproj, NB, HD, HD);
        ln_h_kernel<<<ROWS, 256, 0, stream>>>(x_f, lnp_s + l * HD, lnp_b + l * HD,
                                              gproj, h_h);
        gemm(h_h, WlWt_h + wo, nullptr, nullptr, hl_f, ROWS, HD, HD);
        gemm(h_h, WrWt_h + wo, nullptr, nullptr, hr_f, ROWS, HD, HD);
        attn_kernel<<<ROWS, 256, 0, stream>>>(hl_f, hr_f, adj, Wa + l * NH * NF, hg_h);
        gemm(hg_h, opWt_h + wo, op_b + l * HD, x_f, x_f, ROWS, HD, HD);
    }

    // ---- readout + heads ----
    mean_kernel<<<NB, 256, 0, stream>>>(x_f, ge_f);
    heads_kernel<<<NB, 256, 0, stream>>>(ge_f, card_W, card_b, at_W, at_b,
                                         src_W, src_b, tgt_W, tgt_b,
                                         v1_W, v1_b, v2_W, v2_b, out);
}
